// gaudi_MambaMixer_73340861546917
// MI455X (gfx1250) — compile-verified, matrix operand order checked by reading, not run on hardware
//
#include <hip/hip_runtime.h>
#include <hip/hip_bf16.h>

// ---------------------------------------------------------------------------
// Mamba mixer for MI455X (gfx1250, wave32, WMMA).
// GEMMs: fp32 -> bf16 tiles in double-buffered LDS, v_wmma_f32_16x16x32_bf16,
// f32 accumulation. Block tile 64(M) x 256(N), 8 waves as 2x4, wave tile
// 32x64 (2x4 wmma). All shapes are template constants so addressing folds
// into immediate offsets. Scan / conv / gating: fp32 VALU kernels.
// ---------------------------------------------------------------------------

typedef __attribute__((ext_vector_type(16))) __bf16 v16bf;
typedef __attribute__((ext_vector_type(8)))  __bf16 v8bf;
typedef __attribute__((ext_vector_type(8)))  float  v8f;

// problem constants
#define CB 2
#define CL 1024
#define CH 2048
#define CI 4096
#define CN 16
#define CR 128
#define CK 4
#define CM (CB * CL)          // 2048 rows for all GEMMs
#define PDIM (2 * CI)         // 8192
#define SSMW (CR + 2 * CN)    // 160

// GEMM tiling
#define TILE_M 64
#define TILE_N 256
#define TILE_K 32
#define LDSS   40             // bf16 row stride (80 B, keeps 16B alignment)

__device__ __forceinline__ v8bf pack_bf16x8(float4 a, float4 b) {
  v8bf r;
  r[0] = (__bf16)a.x; r[1] = (__bf16)a.y; r[2] = (__bf16)a.z; r[3] = (__bf16)a.w;
  r[4] = (__bf16)b.x; r[5] = (__bf16)b.y; r[6] = (__bf16)b.z; r[7] = (__bf16)b.w;
  return r;
}

// N, K, LDA compile-time; GUARD: N-edge guarding (x_proj only, N=160)
// EPI: 0 = none, 1 = softplus(x + bias[col])
template <int N, int K, int LDA, bool GUARD, int EPI>
__global__ __launch_bounds__(256)
void gemm_wmma_bf16(const float* __restrict__ A,
                    const float* __restrict__ W,          // [N,K] row-major
                    float* __restrict__ C,                // [M,N] row-major
                    const float* __restrict__ bias) {
  __shared__ __align__(16) __bf16 lA[2][TILE_M * LDSS];
  __shared__ __align__(16) __bf16 lB[2][TILE_N * LDSS];

  const int tid  = threadIdx.x;
  const int wave = tid >> 5;
  const int lane = tid & 31;
  const int wm   = wave >> 2;        // 0..1 (M)
  const int wn   = wave & 3;         // 0..3 (N), 64 cols each
  const int m0   = blockIdx.x * TILE_M;
  const int n0   = blockIdx.y * TILE_N;
  const int lrow  = lane & 15;
  const int khalf = lane >> 4;

  // --- A tile fill mapping: row tid/4, 8 contiguous cols ---
  const int rA = tid >> 2, cA = (tid & 3) * 8;
  const float* aPtr = A + (size_t)(m0 + rA) * LDA + cA;

  // --- B tile fill mapping: rows tid/2 and tid/2+128, 16 contiguous cols ---
  const int rB = tid >> 1, cB = (tid & 1) * 16;
  int n1 = n0 + rB, n2 = n0 + rB + 128;
  bool val1 = true, val2 = true;
  if (GUARD) {
    val1 = n1 < N; val2 = n2 < N;
    if (!val1) n1 = 0;
    if (!val2) n2 = 0;
  }
  const float* bPtr1 = W + (size_t)n1 * K + cB;
  const float* bPtr2 = W + (size_t)n2 * K + cB;

  v8f acc[2][4] = {};

  int buf = 0;
  for (int k0 = 0; k0 < K; k0 += TILE_K, buf ^= 1) {
    // global loads: 2x b128 for A row chunk, 8x b128 for two B row chunks
    float4 a0 = ((const float4*)aPtr)[0];
    float4 a1 = ((const float4*)aPtr)[1];
    float4 b10 = ((const float4*)bPtr1)[0];
    float4 b11 = ((const float4*)bPtr1)[1];
    float4 b12 = ((const float4*)bPtr1)[2];
    float4 b13 = ((const float4*)bPtr1)[3];
    float4 b20 = ((const float4*)bPtr2)[0];
    float4 b21 = ((const float4*)bPtr2)[1];
    float4 b22 = ((const float4*)bPtr2)[2];
    float4 b23 = ((const float4*)bPtr2)[3];
    if (GUARD) {
      if (!val1) b10 = b11 = b12 = b13 = make_float4(0.f, 0.f, 0.f, 0.f);
      if (!val2) b20 = b21 = b22 = b23 = make_float4(0.f, 0.f, 0.f, 0.f);
    }
    if (k0 + TILE_K < K) {   // cover next tile's HBM latency
      __builtin_prefetch(aPtr + TILE_K, 0, 1);
      __builtin_prefetch(bPtr1 + TILE_K, 0, 1);
      __builtin_prefetch(bPtr2 + TILE_K, 0, 1);
    }
    aPtr += TILE_K; bPtr1 += TILE_K; bPtr2 += TILE_K;

    // convert + pack -> 16B LDS stores into buffer `buf`
    *(v8bf*)&lA[buf][rA * LDSS + cA]            = pack_bf16x8(a0, a1);
    *(v8bf*)&lB[buf][rB * LDSS + cB]            = pack_bf16x8(b10, b11);
    *(v8bf*)&lB[buf][rB * LDSS + cB + 8]        = pack_bf16x8(b12, b13);
    *(v8bf*)&lB[buf][(rB + 128) * LDSS + cB]    = pack_bf16x8(b20, b21);
    *(v8bf*)&lB[buf][(rB + 128) * LDSS + cB + 8]= pack_bf16x8(b22, b23);
    __syncthreads();   // single barrier per step (double-buffered)

    // fragment loads: two ds_read_b128 each, per CDNA5 16-bit layout
    // (lanes 0-15: K 0..7 / 16..23, lanes 16-31: K 8..15 / 24..31)
    v16bf afrag[2], bfrag[4];
#pragma unroll
    for (int mi = 0; mi < 2; ++mi) {
      const __bf16* base = &lA[buf][(wm * 32 + mi * 16 + lrow) * LDSS];
      union { v16bf v; v8bf h[2]; } u;
      u.h[0] = *(const v8bf*)(base + 8 * khalf);
      u.h[1] = *(const v8bf*)(base + 16 + 8 * khalf);
      afrag[mi] = u.v;
    }
#pragma unroll
    for (int ni = 0; ni < 4; ++ni) {
      const __bf16* base = &lB[buf][(wn * 64 + ni * 16 + lrow) * LDSS];
      union { v16bf v; v8bf h[2]; } u;
      u.h[0] = *(const v8bf*)(base + 8 * khalf);
      u.h[1] = *(const v8bf*)(base + 16 + 8 * khalf);
      bfrag[ni] = u.v;
    }

#pragma unroll
    for (int mi = 0; mi < 2; ++mi)
#pragma unroll
      for (int ni = 0; ni < 4; ++ni)
        acc[mi][ni] = __builtin_amdgcn_wmma_f32_16x16x32_bf16(
            false, afrag[mi], false, bfrag[ni],
            (short)0, acc[mi][ni], false, false);
  }

  // store: lane l, vgpr v -> row = v + 8*(l>>4), col = l&15 (coalesced cols).
  // Single base pointer + compile-time offsets (N is a template constant).
  float* cBase = C + (size_t)(m0 + wm * 32 + 8 * khalf) * N
                   + (n0 + wn * 64 + lrow);
#pragma unroll
  for (int mi = 0; mi < 2; ++mi) {
#pragma unroll
    for (int ni = 0; ni < 4; ++ni) {
      if (GUARD && (n0 + wn * 64 + ni * 16 + lrow) >= N) continue;
      float bcol = (EPI == 1) ? bias[n0 + wn * 64 + ni * 16 + lrow] : 0.0f;
#pragma unroll
      for (int v = 0; v < 8; ++v) {
        float val = acc[mi][ni][v];
        if (EPI == 1) {
          float x = val + bcol;
          val = (x > 20.0f) ? x : log1pf(__expf(x));
        }
        cBase[(size_t)(mi * 16 + v) * N + ni * 16] = val;
      }
    }
  }
}

// causal depthwise conv1d (K=4, left zero-pad) + bias + SiLU, 4 channels/thread
__global__ __launch_bounds__(256)
void conv_silu_kernel(const float* __restrict__ proj,
                      const float* __restrict__ conv_w,
                      const float* __restrict__ conv_b,
                      float* __restrict__ u) {
  int idx = blockIdx.x * blockDim.x + threadIdx.x;    // over B*L*(I/4)
  int i4 = (idx & (CI / 4 - 1)) * 4;
  int bl = idx >> 10;                                 // b*L + l
  int l  = bl & (CL - 1);
  float wv[4][4];
#pragma unroll
  for (int c = 0; c < 4; ++c)
    *(float4*)wv[c] = *(const float4*)(conv_w + (size_t)(i4 + c) * CK);
  float4 acc = *(const float4*)(conv_b + i4);
#pragma unroll
  for (int k = 0; k < CK; ++k) {
    if (l - (CK - 1) + k >= 0) {
      float4 p = *(const float4*)(proj + (size_t)(bl - (CK - 1) + k) * PDIM + i4);
      acc.x += wv[0][k] * p.x;
      acc.y += wv[1][k] * p.y;
      acc.z += wv[2][k] * p.z;
      acc.w += wv[3][k] * p.w;
    }
  }
  float4 r;
  r.x = acc.x / (1.0f + __expf(-acc.x));
  r.y = acc.y / (1.0f + __expf(-acc.y));
  r.z = acc.z / (1.0f + __expf(-acc.z));
  r.w = acc.w / (1.0f + __expf(-acc.w));
  *(float4*)(u + (size_t)bl * CI + i4) = r;
}

// selective scan + D skip + SiLU gating. One thread per (b, i), 16 states
// in registers, sequential over L. y overwrites the dt buffer (same-thread
// read-then-write per element, deterministic).
__global__ __launch_bounds__(256)
void scan_kernel(float* __restrict__ dty,             // in: dt, out: gated y
                 const float* __restrict__ u,
                 const float* __restrict__ ssm,       // [M, 160]: ts|Bm|Cm
                 const float* __restrict__ proj,      // gate = proj[:, CI + i]
                 const float* __restrict__ A_log,
                 const float* __restrict__ Dp) {
  int idx = blockIdx.x * blockDim.x + threadIdx.x;    // over B*I
  int i = idx & (CI - 1);
  int b = idx >> 12;
  float Arow[CN], h[CN];
#pragma unroll
  for (int n = 0; n < CN; ++n) {
    Arow[n] = -__expf(A_log[i * CN + n]);
    h[n] = 0.0f;
  }
  float Di = Dp[i];
  for (int l = 0; l < CL; ++l) {
    int m = b * CL + l;
    float dtv = dty[(size_t)m * CI + i];
    float uv  = u[(size_t)m * CI + i];
    float du  = dtv * uv;
    // Bm|Cm: 32 contiguous floats -> 8x b128 loads (uniform per block)
    float bc[32];
    const float4* bc4 = (const float4*)(ssm + (size_t)m * SSMW + CR);
#pragma unroll
    for (int q = 0; q < 8; ++q) ((float4*)bc)[q] = bc4[q];
    float accv = 0.0f;
#pragma unroll
    for (int n = 0; n < CN; ++n) {
      float dA = __expf(dtv * Arow[n]);
      h[n] = dA * h[n] + du * bc[n];
      accv += h[n] * bc[CN + n];
    }
    float g  = proj[(size_t)m * PDIM + CI + i];
    float yv = (accv + Di * uv) * (g / (1.0f + __expf(-g)));
    dty[(size_t)m * CI + i] = yv;
  }
}

extern "C" void kernel_launch(void* const* d_in, const int* in_sizes, int n_in,
                              void* d_out, int out_size, void* d_ws, size_t ws_size,
                              hipStream_t stream) {
  const float* x          = (const float*)d_in[0];
  const float* in_proj_w  = (const float*)d_in[1];
  const float* conv_w     = (const float*)d_in[2];
  const float* conv_b     = (const float*)d_in[3];
  const float* x_proj_w   = (const float*)d_in[4];
  const float* dt_proj_w  = (const float*)d_in[5];
  const float* dt_proj_b  = (const float*)d_in[6];
  const float* A_log      = (const float*)d_in[7];
  const float* Dp         = (const float*)d_in[8];
  const float* out_proj_w = (const float*)d_in[9];
  float* out = (float*)d_out;

  float* proj = (float*)d_ws;                         // [2048, 8192]  64 MB
  float* u    = proj + (size_t)CM * PDIM;             // [2048, 4096]  32 MB
  float* ssm  = u    + (size_t)CM * CI;               // [2048, 160]  1.25 MB
  float* dty  = ssm  + (size_t)CM * SSMW;             // [2048, 4096]  32 MB

  dim3 blk(256);

  // 1) in_proj: proj = x @ in_proj_w^T   (M=2048, N=8192, K=2048)
  gemm_wmma_bf16<PDIM, CH, CH, false, 0>
      <<<dim3(CM / TILE_M, PDIM / TILE_N), blk, 0, stream>>>(
      x, in_proj_w, proj, nullptr);

  // 2) depthwise conv + SiLU -> u (stored [B*L, I], i.e. uT)
  conv_silu_kernel<<<(CM * (CI / 4)) / 256, blk, 0, stream>>>(proj, conv_w, conv_b, u);

  // 3) x_proj: ssm = u @ x_proj_w^T     (M=2048, N=160, K=4096) — guarded
  gemm_wmma_bf16<SSMW, CI, CI, true, 0>
      <<<dim3(CM / TILE_M, 1), blk, 0, stream>>>(
      u, x_proj_w, ssm, nullptr);

  // 4) dt_proj + fused softplus(.+b): dt = softplus(ssm[:, :128] @ W^T + b)
  gemm_wmma_bf16<CI, CR, SSMW, false, 1>
      <<<dim3(CM / TILE_M, CI / TILE_N), blk, 0, stream>>>(
      ssm, dt_proj_w, dty, dt_proj_b);

  // 5) selective scan + gating: y (into dty)
  scan_kernel<<<(CB * CI) / 256, blk, 0, stream>>>(dty, u, ssm, proj, A_log, Dp);

  // 6) out_proj: out = y @ out_proj_w^T  (M=2048, N=2048, K=4096)
  gemm_wmma_bf16<CH, CI, CI, false, 0>
      <<<dim3(CM / TILE_M, CH / TILE_N), blk, 0, stream>>>(
      dty, out_proj_w, out, nullptr);
}
